// Encoder_49873160241154
// MI455X (gfx1250) — compile-verified, matrix operand order checked by reading, not run on hardware
//
#include <hip/hip_runtime.h>
#include <math.h>

// ---------------------------------------------------------------------------
// CDNA5 (gfx1250) encoder pipeline: bf16 WMMA GEMMs + fused elementwise ops.
// D = 2048, B = 64, IMAGES = 20, NEIGHBOR = 5, T = 16.
// GEMMs: v_wmma_f32_16x16x32_bf16, 64x64 register tile per wave (16 v8f accs),
// tiles staged to LDS with global_load_async_to_lds_b128 (ASYNCcnt path).
// ---------------------------------------------------------------------------

typedef __attribute__((ext_vector_type(16))) __bf16 v16bf;
typedef __attribute__((ext_vector_type(8)))  float  v8f;

#define D_DIM   2048
#define SCALE_A 0.022097086912079608f   // 1/sqrt(2048)

union FragBF {
  v16bf v;
  uint4 q[2];
};

// ---------------------------------------------------------------------------
// GEMM: Cout[M x N] = A[M x K](bf16, row stride lda) @ W[N x K]^T (bf16)
//                     + bias[N] (opt) + Cadd[M x N] (opt, row stride ldadd)
// Block: 256 threads (8 waves, wave32), tile 128x256, BK=64, double-buffered
// LDS filled by async-to-LDS copies; each wave computes a 64x64 output patch.
// ---------------------------------------------------------------------------
#define BM 128
#define BN 256
#define BK 64
#define LDT 72   // LDS row pitch in bf16 (64 + 8) -> conflict-free b128 frag loads

__global__ __launch_bounds__(256)
void gemm_bf16_wmma(const __bf16* __restrict__ A, long lda, int M,
                    const __bf16* __restrict__ W, long ldw,
                    const float* __restrict__ bias,
                    const float* __restrict__ Cadd, long ldadd,
                    float* __restrict__ Cout, long ldc,
                    int K)
{
  __shared__ __bf16 smA[2][BM * LDT];
  __shared__ __bf16 smB[2][BN * LDT];

  const int tid   = threadIdx.x;
  const int wave  = tid >> 5;
  const int lane  = tid & 31;
  const int tileN = blockIdx.x * BN;
  const int tileM = blockIdx.y * BM;
  const int wm = wave >> 2;     // 0..1 : 64-row band of the 128-row tile
  const int wn = wave & 3;      // 0..3 : 64-col band of the 256-col tile
  const int l16 = lane & 15;
  const int khi = lane >> 4;    // selects K chunk {0..7,16..23} vs {8..15,24..31}

  v8f acc[4][4];
  v8f zero;
  #pragma unroll
  for (int e = 0; e < 8; ++e) zero[e] = 0.0f;
  #pragma unroll
  for (int i = 0; i < 4; ++i)
    #pragma unroll
    for (int j = 0; j < 4; ++j) acc[i][j] = zero;

  const int r0 = tid >> 3;   // 0..31 (row within a 32-row slab)
  const int ch = tid & 7;    // 8-half (16B) chunk within a 64-half row

  // LDS byte offsets of the staging buffers (low 32 bits of the flat address
  // of an LDS object are its offset within the workgroup's LDS allocation).
  const unsigned ldsA = (unsigned)(size_t)&smA[0][0];
  const unsigned ldsB = (unsigned)(size_t)&smB[0][0];

  auto load_tile = [&](int buf, int k0) {
    const unsigned baseA = ldsA + (unsigned)buf * (BM * LDT * 2);
    const unsigned baseB = ldsB + (unsigned)buf * (BN * LDT * 2);
    #pragma unroll
    for (int p = 0; p < 4; ++p) {          // A: 128 rows x 64 halves
      const int r = p * 32 + r0;
      int ra = tileM + r; if (ra > M - 1) ra = M - 1;   // clamp partial M tile
      const __bf16* g = A + (long)ra * lda + k0 + ch * 8;
      const unsigned l = baseA + (unsigned)(r * (LDT * 2) + ch * 16);
      asm volatile("global_load_async_to_lds_b128 %0, %1, off"
                   :: "v"(l), "v"(g) : "memory");
    }
    #pragma unroll
    for (int p = 0; p < 8; ++p) {          // B: 256 rows x 64 halves
      const int r = p * 32 + r0;
      const __bf16* g = W + (long)(tileN + r) * ldw + k0 + ch * 8;
      const unsigned l = baseB + (unsigned)(r * (LDT * 2) + ch * 16);
      asm volatile("global_load_async_to_lds_b128 %0, %1, off"
                   :: "v"(l), "v"(g) : "memory");
    }
  };

  const int nk = K / BK;
  load_tile(0, 0);

  for (int kt = 0; kt < nk; ++kt) {
    // Own async copies for the buffer we are about to read must have landed;
    // barrier makes every wave's copies + previous-tile LDS reads globally done.
    asm volatile("s_wait_asynccnt 0x0" ::: "memory");
    __syncthreads();
    if (kt + 1 < nk) load_tile((kt + 1) & 1, (kt + 1) * BK);

    const __bf16* sA = smA[kt & 1];
    const __bf16* sB = smB[kt & 1];
    #pragma unroll
    for (int kk = 0; kk < 2; ++kk) {
      FragBF af[4], bfr[4];
      const int c0 = kk * 32 + khi * 8;
      #pragma unroll
      for (int mi = 0; mi < 4; ++mi) {
        const int row = wm * 64 + mi * 16 + l16;
        af[mi].q[0] = *reinterpret_cast<const uint4*>(&sA[row * LDT + c0]);
        af[mi].q[1] = *reinterpret_cast<const uint4*>(&sA[row * LDT + c0 + 16]);
      }
      #pragma unroll
      for (int ni = 0; ni < 4; ++ni) {
        const int row = wn * 64 + ni * 16 + l16;
        bfr[ni].q[0] = *reinterpret_cast<const uint4*>(&sB[row * LDT + c0]);
        bfr[ni].q[1] = *reinterpret_cast<const uint4*>(&sB[row * LDT + c0 + 16]);
      }
      #pragma unroll
      for (int mi = 0; mi < 4; ++mi)
        #pragma unroll
        for (int ni = 0; ni < 4; ++ni)
          acc[mi][ni] = __builtin_amdgcn_wmma_f32_16x16x32_bf16(
              false, af[mi].v, false, bfr[ni].v, (short)0, acc[mi][ni], false, false);
    }
  }

  // Epilogue: C/D layout — lane = column (n = lane&15), VGPR r = row (m = r + 8*(lane>>4))
  #pragma unroll
  for (int mi = 0; mi < 4; ++mi) {
    #pragma unroll
    for (int ni = 0; ni < 4; ++ni) {
      const int col = tileN + wn * 64 + ni * 16 + l16;
      const float bv = bias ? bias[col] : 0.0f;
      #pragma unroll
      for (int r = 0; r < 8; ++r) {
        const int row = tileM + wm * 64 + mi * 16 + khi * 8 + r;
        if (row < M) {
          float val = acc[mi][ni][r] + bv;
          if (Cadd) val += Cadd[(long)row * ldadd + col];
          Cout[(long)row * ldc + col] = val;
        }
      }
    }
  }
}

// ---------------------------------------------------------------------------
// Windowed attention: per block = one window of L rows (L = 5 or 16).
// dot -> softmax -> weighted sum; emits bf16 result for the next GEMM.
// ---------------------------------------------------------------------------
__global__ __launch_bounds__(256)
void attention_kernel(const float* __restrict__ q, const float* __restrict__ k,
                      const float* __restrict__ v, __bf16* __restrict__ out, int L)
{
  __shared__ float sw[16][17];
  const int tid = threadIdx.x;
  const long base = (long)blockIdx.x * L * D_DIM;

  if (tid < L * L) {
    const int l = tid / L, m = tid % L;
    const float4* qp = reinterpret_cast<const float4*>(q + base + (long)l * D_DIM);
    const float4* kp = reinterpret_cast<const float4*>(k + base + (long)m * D_DIM);
    float acc = 0.0f;
    for (int d4 = 0; d4 < D_DIM / 4; ++d4) {
      const float4 a = qp[d4], b = kp[d4];
      acc += a.x * b.x + a.y * b.y + a.z * b.z + a.w * b.w;
    }
    sw[l][m] = acc * SCALE_A;
  }
  __syncthreads();
  if (tid < L) {
    float mx = -3.4e38f;
    for (int m = 0; m < L; ++m) mx = fmaxf(mx, sw[tid][m]);
    float s = 0.0f;
    for (int m = 0; m < L; ++m) { const float e = __expf(sw[tid][m] - mx); sw[tid][m] = e; s += e; }
    const float inv = 1.0f / s;
    for (int m = 0; m < L; ++m) sw[tid][m] *= inv;
  }
  __syncthreads();
  for (int d = tid; d < D_DIM; d += 256) {
    float acc[16];
    #pragma unroll
    for (int l = 0; l < 16; ++l) acc[l] = 0.0f;
    for (int m = 0; m < L; ++m) {
      const float vv = v[base + (long)m * D_DIM + d];
      #pragma unroll
      for (int l = 0; l < 16; ++l) if (l < L) acc[l] += sw[l][m] * vv;
    }
    for (int l = 0; l < L; ++l) out[base + (long)l * D_DIM + d] = (__bf16)acc[l];
  }
}

// ---------------------------------------------------------------------------
// Fused LSTM gates: gates[M x 8192] -> (h, c); also emits bf16 h for next step.
// ---------------------------------------------------------------------------
__global__ __launch_bounds__(256)
void lstm_gate_kernel(const float* __restrict__ gates,
                      float* __restrict__ c, float* __restrict__ hdst,
                      __bf16* __restrict__ hb, int M, int czero)
{
  const long idx = (long)blockIdx.x * 256 + threadIdx.x;
  if (idx >= (long)M * D_DIM) return;
  const long row = idx >> 11;
  const int  d   = (int)(idx & (D_DIM - 1));
  const float* g = gates + row * (4L * D_DIM);
  const float gi = g[d], gf = g[D_DIM + d], gg = g[2 * D_DIM + d], go = g[3 * D_DIM + d];
  const float si = 1.0f / (1.0f + __expf(-gi));
  const float sf = 1.0f / (1.0f + __expf(-gf));
  const float so = 1.0f / (1.0f + __expf(-go));
  const float cp = czero ? 0.0f : c[idx];
  const float cn = sf * cp + si * tanhf(gg);
  const float hn = so * tanhf(cn);
  c[idx]    = cn;
  hdst[idx] = hn;
  hb[idx]   = (__bf16)hn;
}

// ---------------------------------------------------------------------------
// LayerNorm over D; writes f32 output (new_nsa_feats) + bf16 copy for stage 2.
// ---------------------------------------------------------------------------
__global__ __launch_bounds__(256)
void layernorm_kernel(const float* __restrict__ x, const float* __restrict__ g,
                      const float* __restrict__ b, float* __restrict__ yf,
                      __bf16* __restrict__ yb)
{
  __shared__ float red[256];
  const int tid = threadIdx.x;
  const float* xr = x + (long)blockIdx.x * D_DIM;
  float s = 0.0f;
  for (int d = tid; d < D_DIM; d += 256) s += xr[d];
  red[tid] = s; __syncthreads();
  for (int st = 128; st > 0; st >>= 1) { if (tid < st) red[tid] += red[tid + st]; __syncthreads(); }
  const float mu = red[0] * (1.0f / D_DIM);
  __syncthreads();
  float s2 = 0.0f;
  for (int d = tid; d < D_DIM; d += 256) { const float t = xr[d] - mu; s2 += t * t; }
  red[tid] = s2; __syncthreads();
  for (int st = 128; st > 0; st >>= 1) { if (tid < st) red[tid] += red[tid + st]; __syncthreads(); }
  const float inv = rsqrtf(red[0] * (1.0f / D_DIM) + 1e-5f);
  for (int d = tid; d < D_DIM; d += 256) {
    const float y = (xr[d] - mu) * inv * g[d] + b[d];
    yf[(long)blockIdx.x * D_DIM + d] = y;
    yb[(long)blockIdx.x * D_DIM + d] = (__bf16)y;
  }
}

// Gather sliding windows (idx[t,j] = t+j) from fc_feats and convert to bf16.
__global__ __launch_bounds__(256)
void gather_bf16_kernel(const float* __restrict__ fc, __bf16* __restrict__ xw)
{
  const long idx = (long)blockIdx.x * 256 + threadIdx.x;
  if (idx >= 5120L * D_DIM) return;
  const int  d  = (int)(idx & (D_DIM - 1));
  const long r  = idx >> 11;          // window-row = (b*16 + t)*5 + j
  const int  j  = (int)(r % 5);
  const long wt = r / 5;
  const int  t  = (int)(wt % 16);
  const long bb = wt / 16;
  xw[idx] = (__bf16)fc[((bb * 20) + t + j) * (long)D_DIM + d];
}

__global__ __launch_bounds__(256)
void f2bf_kernel(const float* __restrict__ in, __bf16* __restrict__ out, long n)
{
  const long i = ((long)blockIdx.x * 256 + threadIdx.x) * 4;
  if (i >= n) return;
  const float4 v = *reinterpret_cast<const float4*>(in + i);
  out[i]     = (__bf16)v.x;
  out[i + 1] = (__bf16)v.y;
  out[i + 2] = (__bf16)v.z;
  out[i + 3] = (__bf16)v.w;
}

__global__ __launch_bounds__(256)
void vadd_kernel(const float* __restrict__ a, const float* __restrict__ b,
                 float* __restrict__ o, int n)
{
  const int i = blockIdx.x * 256 + threadIdx.x;
  if (i < n) o[i] = a[i] + b[i];
}

// ---------------------------------------------------------------------------
// Host launcher. Workspace budget ~390 MB (bf16 weights 184.5 MB + activations).
// ---------------------------------------------------------------------------
extern "C" void kernel_launch(void* const* d_in, const int* in_sizes, int n_in,
                              void* d_out, int out_size, void* d_ws, size_t ws_size,
                              hipStream_t stream)
{
  (void)in_sizes; (void)n_in; (void)out_size; (void)ws_size;

  const float* fc   = (const float*)d_in[0];
  const float* nWq  = (const float*)d_in[1];
  const float* nWk  = (const float*)d_in[2];
  const float* nWv  = (const float*)d_in[3];
  const float* nbq  = (const float*)d_in[4];
  const float* nbk  = (const float*)d_in[5];
  const float* nbv  = (const float*)d_in[6];
  const float* nWih = (const float*)d_in[7];
  const float* nWhh = (const float*)d_in[8];
  const float* nbih = (const float*)d_in[9];
  const float* nbhh = (const float*)d_in[10];
  const float* sWq  = (const float*)d_in[11];
  const float* sWk  = (const float*)d_in[12];
  const float* sWv  = (const float*)d_in[13];
  const float* sbq  = (const float*)d_in[14];
  const float* sbk  = (const float*)d_in[15];
  const float* sbv  = (const float*)d_in[16];
  const float* sWih = (const float*)d_in[17];
  const float* sWhh = (const float*)d_in[18];
  const float* sbih = (const float*)d_in[19];
  const float* sbhh = (const float*)d_in[20];
  const float* lng  = (const float*)d_in[21];
  const float* lnb  = (const float*)d_in[22];

  char* ws = (char*)d_ws;
  size_t off = 0;
  auto take = [&](size_t bytes) -> char* {
    char* p = ws + off;
    off += (bytes + 255) & ~(size_t)255;
    return p;
  };

  const long SZ_D2  = 2048L * 2048;   // square weight elems
  const long SZ_4D2 = 8192L * 2048;   // 4D x D weight elems

  // bf16 weights (persist for whole launch)
  __bf16* nWqB  = (__bf16*)take(SZ_D2 * 2);
  __bf16* nWkB  = (__bf16*)take(SZ_D2 * 2);
  __bf16* nWvB  = (__bf16*)take(SZ_D2 * 2);
  __bf16* nWihB = (__bf16*)take(SZ_4D2 * 2);
  __bf16* nWhhB = (__bf16*)take(SZ_4D2 * 2);
  __bf16* sWqB  = (__bf16*)take(SZ_D2 * 2);
  __bf16* sWkB  = (__bf16*)take(SZ_D2 * 2);
  __bf16* sWvB  = (__bf16*)take(SZ_D2 * 2);
  __bf16* sWihB = (__bf16*)take(SZ_4D2 * 2);
  __bf16* sWhhB = (__bf16*)take(SZ_4D2 * 2);
  float*  bsum1 = (float*)take(8192 * 4);
  float*  bsum2 = (float*)take(8192 * 4);

  // activations (regions reused across stages)
  __bf16* xw_att = (__bf16*)take(5120L * 2048 * 2);   // xw_b -> att1_b -> att2_b
  float*  qbuf   = (float*)take(5120L * 2048 * 4);    // q1 -> q2
  float*  kbuf   = (float*)take(5120L * 2048 * 4);    // k1 -> k2
  float*  vbuf   = (float*)take(5120L * 2048 * 4);    // v1 -> v2
  float*  gates  = (float*)take(1024L * 8192 * 4);    // gates1 -> gates2
  float*  hbuf   = (float*)take(1024L * 2048 * 4);    // h1 -> h2
  float*  cbuf   = (float*)take(1024L * 2048 * 4);    // c1 -> c2
  __bf16* hb     = (__bf16*)take(1024L * 2048 * 2);   // bf16 h for recurrent GEMM
  __bf16* x2b    = (__bf16*)take(1024L * 2048 * 2);   // layernorm output (bf16)

  auto conv = [&](const float* src, __bf16* dst, long n) {
    f2bf_kernel<<<dim3((unsigned)(n / 4 / 256)), dim3(256), 0, stream>>>(src, dst, n);
  };
  auto gemm = [&](const __bf16* A, long lda, int M, const __bf16* W, int N,
                  const float* bias, const float* Cadd, long ldadd,
                  float* Cout, long ldc) {
    dim3 grid((unsigned)(N / BN), (unsigned)((M + BM - 1) / BM));
    gemm_bf16_wmma<<<grid, dim3(256), 0, stream>>>(A, lda, M, W, 2048, bias,
                                                   Cadd, ldadd, Cout, ldc, 2048);
  };

  // ---- weight / bias preparation -----------------------------------------
  conv(nWq, nWqB, SZ_D2);   conv(nWk, nWkB, SZ_D2);   conv(nWv, nWvB, SZ_D2);
  conv(nWih, nWihB, SZ_4D2); conv(nWhh, nWhhB, SZ_4D2);
  conv(sWq, sWqB, SZ_D2);   conv(sWk, sWkB, SZ_D2);   conv(sWv, sWvB, SZ_D2);
  conv(sWih, sWihB, SZ_4D2); conv(sWhh, sWhhB, SZ_4D2);
  vadd_kernel<<<dim3(32), dim3(256), 0, stream>>>(nbih, nbhh, bsum1, 8192);
  vadd_kernel<<<dim3(32), dim3(256), 0, stream>>>(sbih, sbhh, bsum2, 8192);

  // ---- stage 1: neighbor self-attention (windows of 5) -------------------
  gather_bf16_kernel<<<dim3((unsigned)(5120L * 2048 / 256)), dim3(256), 0, stream>>>(fc, xw_att);
  gemm(xw_att, 2048, 5120, nWqB, 2048, nbq, nullptr, 0, qbuf, 2048);
  gemm(xw_att, 2048, 5120, nWkB, 2048, nbk, nullptr, 0, kbuf, 2048);
  gemm(xw_att, 2048, 5120, nWvB, 2048, nbv, nullptr, 0, vbuf, 2048);
  attention_kernel<<<dim3(1024), dim3(256), 0, stream>>>(qbuf, kbuf, vbuf, xw_att, 5);

  // ---- stage 1 LSTM: 5 steps, M = 1024 -----------------------------------
  for (int t = 0; t < 5; ++t) {
    gemm(xw_att + (long)t * 2048, 5L * 2048, 1024, nWihB, 8192, bsum1,
         nullptr, 0, gates, 8192);
    if (t > 0)
      gemm(hb, 2048, 1024, nWhhB, 8192, nullptr, gates, 8192, gates, 8192);
    lstm_gate_kernel<<<dim3((unsigned)(1024L * 2048 / 256)), dim3(256), 0, stream>>>(
        gates, cbuf, hbuf, hb, 1024, t == 0 ? 1 : 0);
  }

  // ---- layernorm -> new_nsa_feats (output 2) + bf16 copy for stage 2 -----
  float* sa_out  = (float*)d_out;
  float* nsa_out = sa_out + 64L * 2048;
  layernorm_kernel<<<dim3(1024), dim3(256), 0, stream>>>(hbuf, lng, lnb, nsa_out, x2b);

  // ---- stage 2: self-attention over T = 16 -------------------------------
  gemm(x2b, 2048, 1024, sWqB, 2048, sbq, nullptr, 0, qbuf, 2048);
  gemm(x2b, 2048, 1024, sWkB, 2048, sbk, nullptr, 0, kbuf, 2048);
  gemm(x2b, 2048, 1024, sWvB, 2048, sbv, nullptr, 0, vbuf, 2048);
  attention_kernel<<<dim3(64), dim3(256), 0, stream>>>(qbuf, kbuf, vbuf, xw_att, 16);

  // ---- stage 2 LSTM: 16 steps, M = 64; final h -> sa_feats (output 1) ----
  for (int t = 0; t < 16; ++t) {
    gemm(xw_att + (long)t * 2048, 16L * 2048, 64, sWihB, 8192, bsum2,
         nullptr, 0, gates, 8192);
    if (t > 0)
      gemm(hb, 2048, 64, sWhhB, 8192, nullptr, gates, 8192, gates, 8192);
    float* hdst = (t == 15) ? sa_out : hbuf;
    lstm_gate_kernel<<<dim3((unsigned)(64L * 2048 / 256)), dim3(256), 0, stream>>>(
        gates, cbuf, hdst, hb, 64, t == 0 ? 1 : 0);
  }
}
